// ValueNet_44487271252085
// MI455X (gfx1250) — matrix operand order repaired and checked by reading, not compiled, not run on hardware
//
#include <hip/hip_runtime.h>
#include <math.h>

// ---------------------------------------------------------------------------
// Types for CDNA5 WMMA
// ---------------------------------------------------------------------------
typedef __attribute__((ext_vector_type(16))) __bf16 v16bf;
typedef __attribute__((ext_vector_type(8)))  float  v8f;

#define TPB 256

__device__ __forceinline__ float lrelu02(float x) { return x > 0.0f ? x : 0.2f * x; }

__device__ __forceinline__ unsigned short f2bf(float f) {
    unsigned u = __float_as_uint(f);
    unsigned r = (u + 0x7FFFu + ((u >> 16) & 1u)) >> 16;   // round-to-nearest-even
    return (unsigned short)r;
}

__device__ __forceinline__ void atomicMaxFloat(float* addr, float val) {
    if (val >= 0.0f) atomicMax((int*)addr, __float_as_int(val));
    else             atomicMin((unsigned int*)addr, (unsigned int)__float_as_int(val));
}

// edge e in [0, Etot): first E entries from edge_index, rest are self-loops
__device__ __forceinline__ void edge_sd(const int* ei, int E, int e, int& s, int& d) {
    if (e < E) { s = ei[e]; d = ei[E + e]; }
    else       { s = e - E; d = e - E; }
}

// ---------------------------------------------------------------------------
// Utility kernels
// ---------------------------------------------------------------------------
__global__ void k_fill(float* p, float v, long long n) {
    long long i = (long long)blockIdx.x * blockDim.x + threadIdx.x;
    if (i < n) p[i] = v;
}

__global__ void k_copy(float* dst, const float* src, long long n) {
    long long i = (long long)blockIdx.x * blockDim.x + threadIdx.x;
    if (i < n) dst[i] = src[i];
}

// ---------------------------------------------------------------------------
// Weight packing: f32 W[Kin,Nout] -> bf16 in WMMA B-operand tile layout:
//   Wp[tn][kb][lane][j]  (tn = Nout/16 tiles, kb = Kpad/32 blocks,
//                         lane = half*16+l, j = 0..15)
//   element = W[(kb*32 + half*16 + j), tn*16 + l]   (zero if K >= Kin)
// Each lane of the GEMM then loads its 16 bf16 (32 bytes) contiguously.
// ---------------------------------------------------------------------------
__global__ void k_pack_w(const float* W, unsigned short* Wp, int Kin, int Kpad, int Nout) {
    long long i = (long long)blockIdx.x * blockDim.x + threadIdx.x;
    long long total = (long long)Kpad * Nout;
    if (i >= total) return;
    int j    = (int)(i & 15);
    int lane = (int)((i >> 4) & 31);
    long long blk = i >> 9;                    // tn * nkb + kb
    int nkb = Kpad >> 5;
    int tn = (int)(blk / nkb), kb = (int)(blk % nkb);
    int half = lane >> 4, l = lane & 15;
    int k = kb * 32 + half * 16 + j;
    int c = tn * 16 + l;
    float v = (k < Kin) ? W[(long long)k * Nout + c] : 0.0f;
    Wp[i] = f2bf(v);
}

// f32 [N,F] (+bias, +act) -> bf16 [N,Fpad], zero pad
__global__ void k_f2bf(const float* in, const float* bias, unsigned short* out,
                       int N, int F, int Fpad, int act) {
    long long i = (long long)blockIdx.x * blockDim.x + threadIdx.x;
    if (i >= (long long)N * Fpad) return;
    int n = (int)(i / Fpad), c = (int)(i % Fpad);
    float v = 0.0f;
    if (c < F) {
        v = in[(long long)n * F + c];
        if (bias) v += bias[c];
        if (act == 1) v = lrelu02(v);
        else if (act == 2) v = v > 0.0f ? v : 0.0f;
    }
    out[i] = f2bf(v);
}

// ---------------------------------------------------------------------------
// WMMA GEMM: out[M,Nout] = act(A[M,Kpad](bf16) @ W(packed bf16) + bias)
// one wave per 16x16 output tile; K-loop step 32; M%16==0, Nout%16==0, Kpad%32==0
// A operand: two 16B loads/lane from row-major bf16; B operand: two 16B loads
// from the pre-packed tile layout above.
// ---------------------------------------------------------------------------
__global__ void k_gemm_bf16(const unsigned short* A_, const unsigned short* Wp_,
                            const float* bias, float* out,
                            int M, int Kpad, int Nout, int act) {
    const __bf16* A  = reinterpret_cast<const __bf16*>(A_);
    const __bf16* Wp = reinterpret_cast<const __bf16*>(Wp_);
    int wid = (int)(blockIdx.x * (blockDim.x >> 5) + (threadIdx.x >> 5));
    int ntn = Nout >> 4;
    int tiles = (M >> 4) * ntn;
    if (wid >= tiles) return;                       // wave-uniform exit: EXEC stays full
    int tm = wid / ntn, tn = wid % ntn;
    int lane = (int)(threadIdx.x & 31);
    int half = lane >> 4;                           // 0: lanes 0-15, 1: lanes 16-31
    int l    = lane & 15;
    int nkb  = Kpad >> 5;

    v8f acc = {};
    // A operand: lane row m=l; half==0 -> K {k0..k0+15}; half==1 -> K {k0+16..k0+31}
    const __bf16* arow  = A  + (size_t)(tm * 16 + l) * Kpad + half * 16;
    const __bf16* wbase = Wp + ((size_t)tn * nkb << 9) + lane * 16;

    for (int kb = 0; kb < nkb; ++kb) {
        union { v16bf v; uint4 u[2]; } ua;
        const __bf16* ap = arow + (kb << 5);
        ua.u[0] = *reinterpret_cast<const uint4*>(ap);
        ua.u[1] = *reinterpret_cast<const uint4*>(ap + 8);

        union { v16bf v; uint4 u[2]; } ub;
        const __bf16* wp = wbase + ((size_t)kb << 9);
        ub.u[0] = *reinterpret_cast<const uint4*>(wp);
        ub.u[1] = *reinterpret_cast<const uint4*>(wp + 8);

        acc = __builtin_amdgcn_wmma_f32_16x16x32_bf16(false, ua.v, false, ub.v,
                                                      (short)0, acc, false, false);
    }

    int col = tn * 16 + l;
    float bv = bias ? bias[col] : 0.0f;
#pragma unroll
    for (int i = 0; i < 8; ++i) {
        int row = tm * 16 + i + half * 8;           // C/D layout: vgpr i, half -> +8 rows
        float v = acc[i] + bv;
        if (act == 1)      v = lrelu02(v);
        else if (act == 2) v = v > 0.0f ? v : 0.0f;
        out[(size_t)row * Nout + col] = v;
    }
}

// ---------------------------------------------------------------------------
// Input builders
// ---------------------------------------------------------------------------
__global__ void k_build_h0(const float* x, const float* J, const float* sc,
                           unsigned short* out, int N) {
    long long i = (long long)blockIdx.x * blockDim.x + threadIdx.x;
    if (i >= (long long)N * 96) return;
    int n = (int)(i / 96), c = (int)(i % 96);
    float v = 0.0f;
    if (c < 64)       v = x[(long long)n * 64 + c];
    else if (c == 64) v = J[n];
    else if (c == 65) v = sc[n];
    out[i] = f2bf(v);
}

__global__ void k_build_xe(const float* h, const float* J, const float* sv, const float* inf,
                           unsigned short* out, int N) {
    long long i = (long long)blockIdx.x * blockDim.x + threadIdx.x;
    if (i >= (long long)N * 96) return;
    int n = (int)(i / 96), c = (int)(i % 96);
    float v = 0.0f;
    if (c < 64)       v = h[(long long)n * 64 + c];
    else if (c == 64) v = J[n];
    else if (c == 65) v = sv[n];
    else if (c == 66) v = inf[n];
    out[i] = f2bf(v);
}

__global__ void k_build_xs(const float* h, const float* pooled, const int* batch,
                           const float* J, const float* sv, const float* inf, const float* sc,
                           const float* Om, const float* Ph, const float* La,
                           unsigned short* out, int N) {
    long long i = (long long)blockIdx.x * blockDim.x + threadIdx.x;
    if (i >= (long long)N * 288) return;
    int n = (int)(i / 288), c = (int)(i % 288);
    int b = batch[n];
    float v = 0.0f;
    if (c < 64)        v = h[(long long)n * 64 + c];
    else if (c < 256)  v = pooled[b * 192 + (c - 64)];
    else if (c == 256) v = J[n];
    else if (c == 257) v = sv[n];
    else if (c == 258) v = inf[n];
    else if (c == 259) v = sc[n];
    else if (c == 260) v = Om[b];
    else if (c == 261) v = Ph[b];
    else if (c == 262) v = La[b];
    out[i] = f2bf(v);
}

// ---------------------------------------------------------------------------
// GAT attention
// ---------------------------------------------------------------------------
__global__ void k_gat_scores(const float* F, const float* asrc, const float* adst,
                             float* ss, float* sd, int N, int H, int C) {
    long long i = (long long)blockIdx.x * blockDim.x + threadIdx.x;
    if (i >= (long long)N * H) return;
    int n = (int)(i / H), h = (int)(i % H);
    const float* f  = F + (long long)n * H * C + h * C;
    const float* as = asrc + h * C;
    const float* ad = adst + h * C;
    float s1 = 0.0f, s2 = 0.0f;
    for (int c = 0; c < C; ++c) { float v = f[c]; s1 += v * as[c]; s2 += v * ad[c]; }
    ss[i] = s1; sd[i] = s2;
}

__global__ void k_edge_max(const int* ei, int E, int Etot,
                           const float* ss, const float* sd, float* m, int H) {
    long long i = (long long)blockIdx.x * blockDim.x + threadIdx.x;
    if (i >= (long long)Etot * H) return;
    int e = (int)(i / H), h = (int)(i % H);
    int s, d; edge_sd(ei, E, e, s, d);
    float v = lrelu02(ss[s * H + h] + sd[d * H + h]);
    atomicMaxFloat(&m[d * H + h], v);
}

__global__ void k_edge_expsum(const int* ei, int E, int Etot,
                              const float* ss, const float* sd, const float* m,
                              float* sum, float* aex, int H) {
    long long i = (long long)blockIdx.x * blockDim.x + threadIdx.x;
    if (i >= (long long)Etot * H) return;
    int e = (int)(i / H), h = (int)(i % H);
    int s, d; edge_sd(ei, E, e, s, d);
    float v  = lrelu02(ss[s * H + h] + sd[d * H + h]);
    float ex = expf(v - m[d * H + h]);
    aex[(long long)e * H + h] = ex;
    atomicAdd(&sum[d * H + h], ex);
}

__global__ void k_gat_agg(const int* ei, int E, int Etot, const float* F,
                          const float* aex, const float* sum, float* out, int H, int C) {
    long long i = (long long)blockIdx.x * blockDim.x + threadIdx.x;
    if (i >= (long long)Etot * C) return;
    int e = (int)(i / C), c = (int)(i % C);
    int s, d; edge_sd(ei, E, e, s, d);
    for (int h = 0; h < H; ++h) {
        float w = aex[(long long)e * H + h] / (sum[d * H + h] + 1e-16f);
        atomicAdd(&out[(long long)d * H * C + h * C + c],
                  F[(long long)s * H * C + h * C + c] * w);
    }
}

// ---------------------------------------------------------------------------
// APPNP
// ---------------------------------------------------------------------------
__global__ void k_deg(const int* ei, int E, int Etot, float* deg) {
    long long i = (long long)blockIdx.x * blockDim.x + threadIdx.x;
    if (i >= Etot) return;
    int s, d; edge_sd(ei, E, (int)i, s, d);
    atomicAdd(&deg[d], 1.0f);
}

__global__ void k_dinv(float* deg, int N) {
    long long i = (long long)blockIdx.x * blockDim.x + threadIdx.x;
    if (i >= N) return;
    float d = deg[i];
    deg[i] = d > 0.0f ? rsqrtf(d) : 0.0f;
}

__global__ void k_appnp_scatter(const int* ei, int E, int Etot, const float* dinv,
                                const float* h, float* acc) {
    long long i = (long long)blockIdx.x * blockDim.x + threadIdx.x;
    if (i >= (long long)Etot * 64) return;
    int e = (int)(i / 64), c = (int)(i % 64);
    int s, d; edge_sd(ei, E, e, s, d);
    atomicAdd(&acc[(long long)d * 64 + c], h[(long long)s * 64 + c] * dinv[s] * dinv[d]);
}

__global__ void k_appnp_combine(float* acc, const float* x0, long long n) {
    long long i = (long long)blockIdx.x * blockDim.x + threadIdx.x;
    if (i < n) acc[i] = 0.9f * acc[i] + 0.1f * x0[i];
}

// ---------------------------------------------------------------------------
// Pooling + head
// ---------------------------------------------------------------------------
__global__ void k_dot_col(const float* X, const float* w, const float* b,
                          float* g, int N, int K) {
    long long n = (long long)blockIdx.x * blockDim.x + threadIdx.x;
    if (n >= N) return;
    float s = b[0];
    const float* xr = X + n * K;
    for (int k = 0; k < K; ++k) s += xr[k] * w[k];
    g[n] = s;
}

__global__ void k_seg_max(const float* g, const int* batch, float* m, int N) {
    long long n = (long long)blockIdx.x * blockDim.x + threadIdx.x;
    if (n >= N) return;
    atomicMaxFloat(&m[batch[n]], g[n]);
}

__global__ void k_seg_expsum(const float* g, const int* batch, const float* m,
                             float* sum, float* a, int N) {
    long long n = (long long)blockIdx.x * blockDim.x + threadIdx.x;
    if (n >= N) return;
    int b = batch[n];
    float ex = expf(g[n] - m[b]);
    a[n] = ex;
    atomicAdd(&sum[b], ex);
}

__global__ void k_pool_accum(const float* a, const float* sum, const int* batch,
                             const float* V, float* pooled, int N, int poff) {
    long long i = (long long)blockIdx.x * blockDim.x + threadIdx.x;
    if (i >= (long long)N * 64) return;
    int n = (int)(i / 64), c = (int)(i % 64);
    int b = batch[n];
    float w = a[n] / (sum[b] + 1e-16f);
    atomicAdd(&pooled[b * 192 + poff + c], w * V[(long long)n * 64 + c]);
}

__global__ void k_final(const float* S2, const float* w5, const float* b5,
                        const int* batch, float* out, int N) {
    long long n = (long long)blockIdx.x * blockDim.x + threadIdx.x;
    if (n >= N) return;
    float s = b5[0];
    const float* r = S2 + n * 64;
    for (int k = 0; k < 64; ++k) s += r[k] * w5[k];
    s = 1.0f / (1.0f + expf(-s));
    atomicAdd(&out[batch[n]], s);
}

// ---------------------------------------------------------------------------
// Host orchestration
// ---------------------------------------------------------------------------
static inline unsigned gdim(long long n) { return (unsigned)((n + TPB - 1) / TPB); }

extern "C" void kernel_launch(void* const* d_in, const int* in_sizes, int n_in,
                              void* d_out, int out_size, void* d_ws, size_t ws_size,
                              hipStream_t stream) {
    (void)in_sizes; (void)n_in; (void)out_size; (void)ws_size;
    const int N = 20000, E = 320000, ETOT = E + N, B = 16;

    const float* x   = (const float*)d_in[0];
    const int*   ei  = (const int*)  d_in[1];
    const int*   bat = (const int*)  d_in[2];
    const float* Om  = (const float*)d_in[3];
    const float* Ph  = (const float*)d_in[4];
    const float* La  = (const float*)d_in[5];
    const float* J   = (const float*)d_in[6];
    const float* sv  = (const float*)d_in[7];
    const float* inf = (const float*)d_in[8];
    const float* sc  = (const float*)d_in[9];
    // params (JAX pytree: dict keys sorted)
    const float* g1W  = (const float*)d_in[10];
    const float* g1ad = (const float*)d_in[11];
    const float* g1as = (const float*)d_in[12];
    const float* g1b  = (const float*)d_in[13];
    const float* g2W  = (const float*)d_in[14];
    const float* g2ad = (const float*)d_in[15];
    const float* g2as = (const float*)d_in[16];
    const float* g2b  = (const float*)d_in[17];
    const float* l1W  = (const float*)d_in[18];
    const float* l1b  = (const float*)d_in[19];
    const float* l2W  = (const float*)d_in[20];
    const float* l2b  = (const float*)d_in[21];
    const float* l3W  = (const float*)d_in[22];
    const float* l3b  = (const float*)d_in[23];
    const float* l4W  = (const float*)d_in[24];
    const float* l4b  = (const float*)d_in[25];
    const float* l5W  = (const float*)d_in[26];
    const float* l5b  = (const float*)d_in[27];
    const float *pg1W[3], *pg1b[3], *pg2W[3], *pg2b[3], *pn1W[3], *pn1b[3], *pn2W[3], *pn2b[3];
    for (int p = 0; p < 3; ++p) {
        int base = 28 + p * 8;
        pg1W[p] = (const float*)d_in[base + 0]; pg1b[p] = (const float*)d_in[base + 1];
        pg2W[p] = (const float*)d_in[base + 2]; pg2b[p] = (const float*)d_in[base + 3];
        pn1W[p] = (const float*)d_in[base + 4]; pn1b[p] = (const float*)d_in[base + 5];
        pn2W[p] = (const float*)d_in[base + 6]; pn2b[p] = (const float*)d_in[base + 7];
    }

    // -------- workspace bump allocation --------
    char* wp = (char*)d_ws;
    auto take = [&](size_t bytes) -> void* {
        void* r = wp; wp += (bytes + 255) & ~(size_t)255; return r;
    };
    unsigned short* stageA = (unsigned short*)take((size_t)N * 512 * 2);
    unsigned short* stageB = (unsigned short*)take((size_t)N * 128 * 2);
    float* FA   = (float*)take((size_t)N * 512 * 4);
    float* FB   = (float*)take((size_t)N * 512 * 4);
    float* X0   = (float*)take((size_t)N * 64 * 4);
    float* Hc   = (float*)take((size_t)N * 64 * 4);
    float* Ha   = (float*)take((size_t)N * 64 * 4);
    float* ssrc = (float*)take((size_t)N * 4 * 4);
    float* sdst = (float*)take((size_t)N * 4 * 4);
    float* mbuf = (float*)take((size_t)N * 4 * 4);
    float* sbuf = (float*)take((size_t)N * 4 * 4);
    float* aex  = (float*)take((size_t)ETOT * 4 * 4);
    float* dinv = (float*)take((size_t)N * 4);
    float* gbuf = (float*)take((size_t)N * 4);
    float* abuf = (float*)take((size_t)N * 4);
    float* segm = (float*)take(64);
    float* segs = (float*)take(64);
    float* pooled = (float*)take((size_t)B * 192 * 4);
    unsigned short* Wg1 = (unsigned short*)take((size_t)96 * 512 * 2);
    unsigned short* Wl1 = (unsigned short*)take((size_t)512 * 128 * 2);
    unsigned short* Wg2 = (unsigned short*)take((size_t)128 * 256 * 2);
    unsigned short* Wl2 = (unsigned short*)take((size_t)256 * 64 * 2);
    unsigned short* Wl3 = (unsigned short*)take((size_t)288 * 128 * 2);
    unsigned short* Wl4 = (unsigned short*)take((size_t)128 * 64 * 2);
    unsigned short *Wpg1[3], *Wpn1[3], *Wpn2[3];
    for (int p = 0; p < 3; ++p) {
        Wpg1[p] = (unsigned short*)take((size_t)96 * 128 * 2);
        Wpn1[p] = (unsigned short*)take((size_t)96 * 128 * 2);
        Wpn2[p] = (unsigned short*)take((size_t)128 * 64 * 2);
    }

    auto gemm = [&](const unsigned short* A, const unsigned short* W, const float* bias,
                    float* out, int M, int Kpad, int Nout, int act) {
        int tiles = (M / 16) * (Nout / 16);
        unsigned grid = (unsigned)((tiles + 7) / 8);  // 8 waves / 256-thread block
        k_gemm_bf16<<<grid, TPB, 0, stream>>>(A, W, bias, out, M, Kpad, Nout, act);
    };
    auto packw = [&](const float* W, unsigned short* Wp, int Kin, int Kpad, int Nout) {
        long long n = (long long)Kpad * Nout;
        k_pack_w<<<gdim(n), TPB, 0, stream>>>(W, Wp, Kin, Kpad, Nout);
    };
    auto fill = [&](float* p, float v, long long n) {
        k_fill<<<gdim(n), TPB, 0, stream>>>(p, v, n);
    };
    const float NEG_INF = -__builtin_huge_valf();

    // -------- pack weights to WMMA B-operand layout (K zero-padded to mult of 32) ----
    packw(g1W, Wg1, 66, 96, 512);
    packw(l1W, Wl1, 512, 512, 128);
    packw(g2W, Wg2, 128, 128, 256);
    packw(l2W, Wl2, 256, 256, 64);
    packw(l3W, Wl3, 263, 288, 128);
    packw(l4W, Wl4, 128, 128, 64);
    for (int p = 0; p < 3; ++p) {
        packw(pg1W[p], Wpg1[p], 67, 96, 128);
        packw(pn1W[p], Wpn1[p], 67, 96, 128);
        packw(pn2W[p], Wpn2[p], 128, 128, 64);
    }

    // -------- GAT layer 1 --------
    k_build_h0<<<gdim((long long)N * 96), TPB, 0, stream>>>(x, J, sc, stageA, N);
    gemm(stageA, Wg1, nullptr, FA, N, 96, 512, 0);                        // h = x@W [N,4,128]
    k_gat_scores<<<gdim((long long)N * 4), TPB, 0, stream>>>(FA, g1as, g1ad, ssrc, sdst, N, 4, 128);
    fill(mbuf, NEG_INF, (long long)N * 4);
    fill(sbuf, 0.0f, (long long)N * 4);
    k_edge_max<<<gdim((long long)ETOT * 4), TPB, 0, stream>>>(ei, E, ETOT, ssrc, sdst, mbuf, 4);
    k_edge_expsum<<<gdim((long long)ETOT * 4), TPB, 0, stream>>>(ei, E, ETOT, ssrc, sdst, mbuf, sbuf, aex, 4);
    fill(FB, 0.0f, (long long)N * 512);
    k_gat_agg<<<gdim((long long)ETOT * 128), TPB, 0, stream>>>(ei, E, ETOT, FA, aex, sbuf, FB, 4, 128);
    k_f2bf<<<gdim((long long)N * 512), TPB, 0, stream>>>(FB, g1b, stageA, N, 512, 512, 0);
    gemm(stageA, Wl1, l1b, FA, N, 512, 128, 1);                           // lrelu(lin1)

    // -------- GAT layer 2 --------
    k_f2bf<<<gdim((long long)N * 128), TPB, 0, stream>>>(FA, nullptr, stageA, N, 128, 128, 0);
    gemm(stageA, Wg2, nullptr, FB, N, 128, 256, 0);                       // h [N,4,64]
    k_gat_scores<<<gdim((long long)N * 4), TPB, 0, stream>>>(FB, g2as, g2ad, ssrc, sdst, N, 4, 64);
    fill(mbuf, NEG_INF, (long long)N * 4);
    fill(sbuf, 0.0f, (long long)N * 4);
    k_edge_max<<<gdim((long long)ETOT * 4), TPB, 0, stream>>>(ei, E, ETOT, ssrc, sdst, mbuf, 4);
    k_edge_expsum<<<gdim((long long)ETOT * 4), TPB, 0, stream>>>(ei, E, ETOT, ssrc, sdst, mbuf, sbuf, aex, 4);
    fill(FA, 0.0f, (long long)N * 256);
    k_gat_agg<<<gdim((long long)ETOT * 64), TPB, 0, stream>>>(ei, E, ETOT, FB, aex, sbuf, FA, 4, 64);
    k_f2bf<<<gdim((long long)N * 256), TPB, 0, stream>>>(FA, g2b, stageA, N, 256, 256, 0);
    gemm(stageA, Wl2, l2b, X0, N, 256, 64, 1);                            // lrelu(lin2) -> x0

    // -------- APPNP (K=10, alpha=0.1) --------
    fill(dinv, 0.0f, N);
    k_deg<<<gdim(ETOT), TPB, 0, stream>>>(ei, E, ETOT, dinv);
    k_dinv<<<gdim(N), TPB, 0, stream>>>(dinv, N);
    k_copy<<<gdim((long long)N * 64), TPB, 0, stream>>>(Hc, X0, (long long)N * 64);
    float* hcur = Hc; float* hacc = Ha;
    for (int t = 0; t < 10; ++t) {
        fill(hacc, 0.0f, (long long)N * 64);
        k_appnp_scatter<<<gdim((long long)ETOT * 64), TPB, 0, stream>>>(ei, E, ETOT, dinv, hcur, hacc);
        k_appnp_combine<<<gdim((long long)N * 64), TPB, 0, stream>>>(hacc, X0, (long long)N * 64);
        float* tmp = hcur; hcur = hacc; hacc = tmp;
    }

    // -------- attention pools --------
    k_build_xe<<<gdim((long long)N * 96), TPB, 0, stream>>>(hcur, J, sv, inf, stageB, N);
    fill(pooled, 0.0f, (long long)B * 192);
    for (int p = 0; p < 3; ++p) {
        gemm(stageB, Wpg1[p], pg1b[p], FA, N, 96, 128, 2);                // relu(gate1)
        k_dot_col<<<gdim(N), TPB, 0, stream>>>(FA, pg2W[p], pg2b[p], gbuf, N, 128);
        fill(segm, NEG_INF, B);
        fill(segs, 0.0f, B);
        k_seg_max<<<gdim(N), TPB, 0, stream>>>(gbuf, bat, segm, N);
        k_seg_expsum<<<gdim(N), TPB, 0, stream>>>(gbuf, bat, segm, segs, abuf, N);
        gemm(stageB, Wpn1[p], pn1b[p], FA, N, 96, 128, 2);                // relu(nn1)
        k_f2bf<<<gdim((long long)N * 128), TPB, 0, stream>>>(FA, nullptr, stageA, N, 128, 128, 0);
        gemm(stageA, Wpn2[p], pn2b[p], FB, N, 128, 64, 0);                // nn2
        k_pool_accum<<<gdim((long long)N * 64), TPB, 0, stream>>>(abuf, segs, bat, FB, pooled, N, p * 64);
    }

    // -------- head MLP --------
    k_build_xs<<<gdim((long long)N * 288), TPB, 0, stream>>>(hcur, pooled, bat, J, sv, inf, sc,
                                                             Om, Ph, La, stageA, N);
    gemm(stageA, Wl3, l3b, FA, N, 288, 128, 1);                           // lrelu(lin3)
    k_f2bf<<<gdim((long long)N * 128), TPB, 0, stream>>>(FA, nullptr, stageB, N, 128, 128, 0);
    gemm(stageB, Wl4, l4b, FB, N, 128, 64, 1);                            // lrelu(lin4)
    fill((float*)d_out, 0.0f, B);
    k_final<<<gdim(N), TPB, 0, stream>>>(FB, l5W, l5b, bat, (float*)d_out, N);
}
